// GAT_73572789781154
// MI455X (gfx1250) — compile-verified, matrix operand order checked by reading, not run on hardware
//
#include <hip/hip_runtime.h>
#include <math.h>

typedef __attribute__((ext_vector_type(16))) _Float16 v16h;
typedef __attribute__((ext_vector_type(8)))  _Float16 v8h;
typedef __attribute__((ext_vector_type(8)))  float    v8f;

#define NNODES 4096
#define NWORDS 128            // 4096 bits / 32
#define NHEADS 8
#define NHID   16
#define NFEAT  6
#define ALPHA  0.2f

__device__ __forceinline__ float leaky_(float x) { return x > 0.f ? x : ALPHA * x; }
__device__ __forceinline__ float elu_(float x)   { return x > 0.f ? x : __expf(x) - 1.f; }

// ---------------------------------------------------------------------------
// Kernel 0: pack adj>0 into a bitmask (4096 x 128 words). One word per wave:
// 32 lanes read 32 consecutive ints (128B coalesced), ballot into one word.
// This is the ONLY full read of the 64 MB adjacency; everything downstream
// touches the 2 MB bitmask, which is L2-resident (192 MB L2).
// ---------------------------------------------------------------------------
__global__ __launch_bounds__(256) void gat_pack(const int* __restrict__ adj,
                                                unsigned int* __restrict__ mb)
{
    int lane = threadIdx.x;                          // 0..31
    int wid  = blockIdx.x * 8 + threadIdx.y;         // word id, 0..524287
    int v = adj[(size_t)wid * 32 + lane];
#if __has_builtin(__builtin_amdgcn_ballot_w32)
    unsigned int word = __builtin_amdgcn_ballot_w32(v != 0);
#else
    unsigned int word = (v != 0) ? (1u << lane) : 0u;
#pragma unroll
    for (int d = 1; d < 32; d <<= 1) word |= __shfl_xor(word, d, 32);
#endif
    if (lane == 0) mb[wid] = word;
}

// ---------------------------------------------------------------------------
// Kernel 1: per-node linear projections for all heads.
// ---------------------------------------------------------------------------
__global__ __launch_bounds__(256) void gat_precompute(
    const float* __restrict__ x, const float* __restrict__ W,
    const float* __restrict__ a, _Float16* __restrict__ whT,
    float* __restrict__ esrc, float* __restrict__ edst)
{
    int n = blockIdx.x * blockDim.x + threadIdx.x;
    if (n >= NNODES) return;
    float xf[NFEAT];
#pragma unroll
    for (int f = 0; f < NFEAT; ++f) xf[f] = x[n * NFEAT + f];

    for (int h = 0; h < NHEADS; ++h) {
        float wh[NHID];
#pragma unroll
        for (int o = 0; o < NHID; ++o) {
            float s = 0.f;
#pragma unroll
            for (int f = 0; f < NFEAT; ++f) s += xf[f] * W[(h * NFEAT + f) * NHID + o];
            wh[o] = s;
        }
        float es = 0.f, ed = 0.f;
#pragma unroll
        for (int o = 0; o < NHID; ++o) {
            es += wh[o] * a[h * 2 * NHID + o];
            ed += wh[o] * a[h * 2 * NHID + NHID + o];
            whT[(size_t)(h * NHID + o) * NNODES + n] = (_Float16)wh[o];
        }
        esrc[h * NNODES + n] = es;
        edst[h * NNODES + n] = ed;
    }
}

// ---------------------------------------------------------------------------
// Kernel 2: gmax[h] = max_m e_dst[h][m]  (safe softmax shift; leaky monotone)
// ---------------------------------------------------------------------------
__global__ __launch_bounds__(256) void gat_gmax(const float* __restrict__ edst,
                                                float* __restrict__ gmax)
{
    __shared__ float red[256];
    int h = blockIdx.x, t = threadIdx.x;
    float m = -1e30f;
    for (int i = t; i < NNODES; i += 256) m = fmaxf(m, edst[h * NNODES + i]);
    red[t] = m;
    __syncthreads();
    for (int s = 128; s > 0; s >>= 1) { if (t < s) red[t] = fmaxf(red[t], red[t + s]); __syncthreads(); }
    if (t == 0) gmax[h] = red[0];
}

// ---------------------------------------------------------------------------
// Kernel 3: fused masked softmax + att@Wh via WMMA.
// Block = (32,8): 8 waves, one head each; 16 output rows per block.
// Entire 16x4096 bitmask block (8 KB) staged in LDS ONCE; no loop barriers.
// ---------------------------------------------------------------------------
__global__ __launch_bounds__(256) void gat_layer1(
    const unsigned int* __restrict__ mb, const _Float16* __restrict__ whT,
    const float* __restrict__ esrc, const float* __restrict__ edst,
    const float* __restrict__ gmax, float* __restrict__ hout)
{
    __shared__ unsigned int mrow[16][NWORDS + 4];  // +4 pad: conflict-free
    __shared__ float denomS[NHEADS][16];

    const int head = threadIdx.y;
    const int lane = threadIdx.x;          // 0..31 (wave32)
    const int rb   = blockIdx.x;           // row block (16 rows)
    const int r    = lane & 15;            // A: row M ; B: col N
    const int bo   = (lane >> 4) * 8;      // k sub-offset per lane half
    const int row  = rb * 16 + r;
    const int tid  = head * 32 + lane;

    // stage 16 rows x 128 mask words (coalesced, one shot)
#pragma unroll
    for (int i = 0; i < 8; ++i) {
        int idx = tid + i * 256;           // 0..2047
        int rr = idx >> 7, ww = idx & (NWORDS - 1);
        mrow[rr][ww] = mb[(size_t)(rb * 16 + rr) * NWORDS + ww];
    }
    __syncthreads();

    const float es      = esrc[head * NNODES + row];
    const float rowbase = leaky_(es + gmax[head]);
    const float* ed     = edst + head * NNODES;
    const _Float16* wb  = whT + (size_t)(head * NHID + r) * NNODES; // B column N=r

    v8f c = {0.f, 0.f, 0.f, 0.f, 0.f, 0.f, 0.f, 0.f};
    float denom = 0.f;

    for (int t = 0; t < NWORDS / 8; ++t) {
#pragma unroll
        for (int sub = 0; sub < 8; ++sub) {
            const int w  = t * 8 + sub;
            const int m0 = w * 32;
            const unsigned int word = mrow[r][w];
            const unsigned int wlo  = word >> bo;         // bits for K=bo+i
            const unsigned int whi  = word >> (16 + bo);  // bits for K=16+bo+i

            v16h av;
#pragma unroll
            for (int i = 0; i < 8; ++i) {
                float e0 = leaky_(es + ed[m0 + bo + i]);
                float p0 = ((wlo >> i) & 1u) ? __expf(e0 - rowbase) : 0.f;
                denom += p0;
                av[i] = (_Float16)p0;
                float e1 = leaky_(es + ed[m0 + 16 + bo + i]);
                float p1 = ((whi >> i) & 1u) ? __expf(e1 - rowbase) : 0.f;
                denom += p1;
                av[8 + i] = (_Float16)p1;
            }
            v8h blo = *(const v8h*)(wb + m0 + bo);
            v8h bhi = *(const v8h*)(wb + m0 + 16 + bo);
            v16h bv;
#pragma unroll
            for (int i = 0; i < 8; ++i) { bv[i] = blo[i]; bv[8 + i] = bhi[i]; }

            c = __builtin_amdgcn_wmma_f32_16x16x32_f16(
                    false, av, false, bv, (short)0, c, false, false);
        }
    }

    // row r is split across lanes r and r+16 -> combine partial denominators
    denom += __shfl_xor(denom, 16, 32);
    if (lane < 16) denomS[head][r] = denom;
    __syncthreads();

    // C/D layout: lane holds col N=r, vgpr j -> row j (lanes<16) / j+8
    const int mbase = (lane < 16) ? 0 : 8;
#pragma unroll
    for (int j = 0; j < 8; ++j) {
        int rm = mbase + j;
        float v = c[j] / denomS[head][rm];
        hout[(size_t)(rb * 16 + rm) * (NHEADS * NHID) + head * NHID + r] = elu_(v);
    }
}

// ---------------------------------------------------------------------------
// Kernel 4: Wh2[n] = h[n][:] . W_out   and  gmax2 = max_m a_out[1]*Wh2[m]
// ---------------------------------------------------------------------------
__global__ __launch_bounds__(512) void gat_l2_prep(
    const float* __restrict__ hbuf, const float* __restrict__ Wout,
    const float* __restrict__ aout, float* __restrict__ s, float* __restrict__ gmax2)
{
    __shared__ __align__(16) float wsh[NHEADS * NHID];
    __shared__ float red[512];
    int t = threadIdx.x;
    if (t < NHEADS * NHID) wsh[t] = Wout[t];
    __syncthreads();
    float a1 = aout[1];
    float mx = -1e30f;
    for (int n = t; n < NNODES; n += 512) {
        const float4* hv = (const float4*)(hbuf + (size_t)n * 128);
        float acc = 0.f;
#pragma unroll
        for (int k4 = 0; k4 < 32; ++k4) {
            float4 h4 = hv[k4];
            float4 w4 = *(const float4*)&wsh[k4 * 4];
            acc += h4.x * w4.x + h4.y * w4.y + h4.z * w4.z + h4.w * w4.w;
        }
        s[n] = acc;
        mx = fmaxf(mx, a1 * acc);
    }
    red[t] = mx;
    __syncthreads();
    for (int st = 256; st > 0; st >>= 1) { if (t < st) red[t] = fmaxf(red[t], red[t + st]); __syncthreads(); }
    if (t == 0) gmax2[0] = red[0];
}

// ---------------------------------------------------------------------------
// Kernel 5: layer-2 fused masked softmax mat-vec (C=1), one wave per row.
// Reads only the L2-resident bitmask + 16 KB s vector.
// ---------------------------------------------------------------------------
__global__ __launch_bounds__(256) void gat_l2_att(
    const unsigned int* __restrict__ mb, const float* __restrict__ s,
    const float* __restrict__ aout, const float* __restrict__ gmax2,
    float* __restrict__ out)
{
    int lane = threadIdx.x;
    int n = blockIdx.x * 8 + threadIdx.y;
    float a0 = aout[0], a1 = aout[1];
    float sn = s[n];
    float base = leaky_(a0 * sn + gmax2[0]);
    float num = 0.f, den = 0.f;
#pragma unroll
    for (int chunk = 0; chunk < NWORDS / 32; ++chunk) {
        int widx = chunk * 32 + lane;
        unsigned int word = mb[(size_t)n * NWORDS + widx];
        int mbase = widx * 32;
#pragma unroll
        for (int j4 = 0; j4 < 8; ++j4) {
            float4 sv = *(const float4*)(s + mbase + j4 * 4);
            if ((word >> (j4 * 4 + 0)) & 1u) { float p = __expf(leaky_(a0 * sn + a1 * sv.x) - base); den += p; num += p * sv.x; }
            if ((word >> (j4 * 4 + 1)) & 1u) { float p = __expf(leaky_(a0 * sn + a1 * sv.y) - base); den += p; num += p * sv.y; }
            if ((word >> (j4 * 4 + 2)) & 1u) { float p = __expf(leaky_(a0 * sn + a1 * sv.z) - base); den += p; num += p * sv.z; }
            if ((word >> (j4 * 4 + 3)) & 1u) { float p = __expf(leaky_(a0 * sn + a1 * sv.w) - base); den += p; num += p * sv.w; }
        }
    }
#pragma unroll
    for (int d = 16; d > 0; d >>= 1) {
        num += __shfl_xor(num, d, 32);
        den += __shfl_xor(den, d, 32);
    }
    if (lane == 0) out[n] = elu_(num / den);
}

// ---------------------------------------------------------------------------
extern "C" void kernel_launch(void* const* d_in, const int* in_sizes, int n_in,
                              void* d_out, int out_size, void* d_ws, size_t ws_size,
                              hipStream_t stream)
{
    const float* flow_x = (const float*)d_in[0];   // (1,4096,6,1)
    const int*   adj    = (const int*)d_in[1];     // (4096,4096)
    const float* W      = (const float*)d_in[2];   // (8,6,16)
    const float* a      = (const float*)d_in[3];   // (8,32)
    const float* Wout   = (const float*)d_in[4];   // (128,1)
    const float* aout   = (const float*)d_in[5];   // (2,)
    float* out = (float*)d_out;                    // (1,4096,1,1)

    char* ws = (char*)d_ws;
    _Float16* whT      = (_Float16*)(ws + 0);            // 1,048,576 B
    float* esrc        = (float*)(ws + 1048576);         //   131,072 B
    float* edst        = (float*)(ws + 1179648);         //   131,072 B
    float* gmax        = (float*)(ws + 1310720);         //       256 B
    float* hbuf        = (float*)(ws + 1310976);         // 2,097,152 B
    float* sbuf        = (float*)(ws + 3408128);         //    16,384 B
    float* gmax2       = (float*)(ws + 3424512);         //       256 B
    unsigned int* mbit = (unsigned int*)(ws + 3424768);  // 2,097,152 B

    gat_pack<<<(NNODES * NWORDS) / 8, dim3(32, 8), 0, stream>>>(adj, mbit);
    gat_precompute<<<NNODES / 256, 256, 0, stream>>>(flow_x, W, a, whT, esrc, edst);
    gat_gmax<<<NHEADS, 256, 0, stream>>>(edst, gmax);
    gat_layer1<<<NNODES / 16, dim3(32, 8), 0, stream>>>(mbit, whT, esrc, edst, gmax, hbuf);
    gat_l2_prep<<<1, 512, 0, stream>>>(hbuf, Wout, aout, sbuf, gmax2);
    gat_l2_att<<<NNODES / 8, dim3(32, 8), 0, stream>>>(mbit, sbuf, aout, gmax2, out);
}
